// TimeModel_57191784513675
// MI455X (gfx1250) — matrix operand order repaired and checked by reading, not compiled, not run on hardware
//
#include <hip/hip_runtime.h>

typedef __attribute__((ext_vector_type(16))) __bf16 v16bf;
typedef __attribute__((ext_vector_type(8)))  __bf16 v8bf;
typedef __attribute__((ext_vector_type(8)))  float  v8f;

#define HIDDEN   1024
#define ROWS     64          // M-rows per workgroup (fits register file, 2 WGs/WGP)
#define STRIDE   1032        // act row stride: 1024 + 8 bf16 pad (16B aligned)
#define XS       136         // feature row stride: 128 + 8 bf16 pad (16B aligned)
#define LN_EPS   1e-5f
#define M_TOTAL  131072      // 16*64*128
#define NTHREADS 512         // 16 waves: 4 (M) x 4 (N)
#define QTILES   16          // N-tiles per wave -> acc[16] = 128 VGPRs (no spills)

__device__ __forceinline__ __bf16 f2bf(float f) {
  unsigned u = __builtin_bit_cast(unsigned, f);
  u = (u + 0x7FFFu + ((u >> 16) & 1u)) >> 16;       // round-to-nearest-even
  unsigned short h = (unsigned short)u;
  return __builtin_bit_cast(__bf16, h);
}
__device__ __forceinline__ float bf2f(__bf16 b) {
  unsigned short h = __builtin_bit_cast(unsigned short, b);
  unsigned u = ((unsigned)h) << 16;
  return __builtin_bit_cast(float, u);
}

// ---------------------------------------------------------------------------
// Weight swizzle: f32 [K,N] row-major -> bf16 WMMA-B tiles.
// Tile (kt,nt) at ((nt*KT + kt)*512); lane l owns 16 contiguous bf16:
// B[kt*32 + (e>=8?16:0) + (l>=16?8:0) + (e&7)][nt*16 + (l&15)], e = 0..15.
// ---------------------------------------------------------------------------
__global__ void swizzle_w(const float* __restrict__ W, __bf16* __restrict__ out,
                          int K, int N) {
  int idx = blockIdx.x * 256 + threadIdx.x;
  if (idx >= K * N) return;
  int tile = idx >> 9;
  int r    = idx & 511;
  int lane = r >> 4, e = r & 15;
  int KT = K >> 5;
  int kt = tile % KT, nt = tile / KT;
  int k = kt * 32 + ((e >> 3) << 4) + ((lane >> 4) << 3) + (e & 7);
  int n = nt * 16 + (lane & 15);
  out[idx] = f2bf(W[k * N + n]);
}

// ---------------------------------------------------------------------------
// One layer, register-resident output: each wave owns a 16-row x 256-col
// panel as 16 f32 accumulator tiles. GEMM reads inBuf (LDS); after a
// barrier the LN+ReLU result is written back into outBuf (may alias inBuf).
// B operand is software-pipelined two tiles deep.
// ---------------------------------------------------------------------------
__device__ __forceinline__ void mlp_layer_reg(
    const __bf16* __restrict__ wsz, const float* __restrict__ bias,
    const float* __restrict__ g,    const float* __restrict__ be,
    const __bf16* inBuf, int inStride, __bf16* outBuf,
    float* rowSum, float* rowSumSq, float* gbuf, float* bebuf,
    int KT, int tid)
{
  // stage g/be into LDS, zero LN accumulators
  for (int i = tid; i < HIDDEN; i += NTHREADS) { gbuf[i] = g[i]; bebuf[i] = be[i]; }
  for (int i = tid; i < ROWS;   i += NTHREADS) { rowSum[i] = 0.0f; rowSumSq[i] = 0.0f; }
  __syncthreads();

  const int lane = tid & 31;
  const int wave = tid >> 5;
  const int wm   = wave & 3;          // 16-row tile (4 of them)
  const int wn   = wave >> 2;         // N quarter (0..3)
  const int half = lane >> 4;         // K-half select (A/B operand layouts)
  const int lm   = lane & 15;         // column within tile / A-row within tile
  const int m0   = wm * 16 + half * 8;
  const int ntBase = wn * QTILES;
  const __bf16* aBase = inBuf + (wm * 16 + lm) * inStride + half * 8;
  const int tileStride = KT << 9;     // elements between consecutive n-tiles

  v8f acc[QTILES];
#pragma unroll
  for (int q = 0; q < QTILES; ++q)
    acc[q] = (v8f){0.f,0.f,0.f,0.f,0.f,0.f,0.f,0.f};

#pragma unroll 1
  for (int kt = 0; kt < KT; ++kt) {
    const __bf16* ap = aBase + kt * 32;
    v8bf lo = *(const v8bf*)(ap);                 // K = kt*32 + half*8 ..+7
    v8bf hi = *(const v8bf*)(ap + 16);            // K = kt*32+16 + half*8 ..
    v16bf a = __builtin_shufflevector(lo, hi, 0,1,2,3,4,5,6,7,8,9,10,11,12,13,14,15);
    const __bf16* wp = wsz + ((ntBase * KT + kt) << 9) + (lane << 4);
    __builtin_prefetch(wp + 512, 0, 1);           // stream next K-step
    // two-deep software pipeline on the B operand
    v16bf bq0 = *(const v16bf*)(wp);
    v16bf bq1 = *(const v16bf*)(wp + tileStride);
#pragma unroll
    for (int q = 0; q < QTILES; ++q) {            // 16 independent WMMA chains
      v16bf bnext = bq1;
      if (q + 2 < QTILES)
        bnext = *(const v16bf*)(wp + (q + 2) * tileStride);
      acc[q] = __builtin_amdgcn_wmma_f32_16x16x32_bf16(
          false, a, false, bq0, (short)0, acc[q], false, false);
      bq0 = bq1;
      bq1 = bnext;
    }
  }

  // bias add + LN partial stats, all in registers
  v8f accSum = {0.f,0.f,0.f,0.f,0.f,0.f,0.f,0.f};
  v8f accSq  = {0.f,0.f,0.f,0.f,0.f,0.f,0.f,0.f};
#pragma unroll
  for (int q = 0; q < QTILES; ++q) {
    const float bv = bias[(ntBase + q) * 16 + lm];
    for (int i = 0; i < 8; ++i) {
      float h = acc[q][i] + bv;
      acc[q][i] = h;
      accSum[i] += h;
      accSq[i]  += h * h;
    }
  }
  // butterfly over the 16 lanes that share this row set (once per layer)
  for (int mask = 1; mask <= 8; mask <<= 1) {
    for (int i = 0; i < 8; ++i) {
      accSum[i] += __shfl_xor(accSum[i], mask, 32);
      accSq[i]  += __shfl_xor(accSq[i],  mask, 32);
    }
  }
  if (lm == 0) {
    for (int i = 0; i < 8; ++i) {
      atomicAdd(&rowSum[m0 + i],   accSum[i]);
      atomicAdd(&rowSumSq[m0 + i], accSq[i]);
    }
  }
  __syncthreads();   // all GEMM reads of inBuf done; stats committed

  if (tid < ROWS) {
    float mean = rowSum[tid] * (1.0f / (float)HIDDEN);
    float var  = rowSumSq[tid] * (1.0f / (float)HIDDEN) - mean * mean;
    rowSum[tid]   = mean;
    rowSumSq[tid] = rsqrtf(var + LN_EPS);
  }
  __syncthreads();

  // normalize + affine + relu straight from f32 registers -> bf16 LDS
  float mean[8], rstd[8];
  for (int i = 0; i < 8; ++i) { mean[i] = rowSum[m0 + i]; rstd[i] = rowSumSq[m0 + i]; }
#pragma unroll
  for (int q = 0; q < QTILES; ++q) {
    const int nGlob = (ntBase + q) * 16 + lm;
    const float gv = gbuf[nGlob], bvv = bebuf[nGlob];
    for (int i = 0; i < 8; ++i) {
      float v = (acc[q][i] - mean[i]) * rstd[i] * gv + bvv;
      outBuf[(m0 + i) * STRIDE + nGlob] = f2bf(fmaxf(v, 0.0f));
    }
  }
  __syncthreads();
}

// ---------------------------------------------------------------------------
// Fused MLP: build pair features -> 3x (GEMM+LN+ReLU) -> head -> masked write
// ---------------------------------------------------------------------------
__global__ __launch_bounds__(NTHREADS, 1) void fused_mlp(
    const int*  __restrict__ time_steps,
    const float* __restrict__ sat,
    const unsigned char* __restrict__ cmask,
    const float* __restrict__ task,
    const unsigned char* __restrict__ tmask,
    const float* __restrict__ te,
    const __bf16* __restrict__ w1s, const float* __restrict__ b1,
    const float* __restrict__ g1,   const float* __restrict__ be1,
    const __bf16* __restrict__ w2s, const float* __restrict__ b2,
    const float* __restrict__ g2,   const float* __restrict__ be2,
    const __bf16* __restrict__ w3s, const float* __restrict__ b3,
    const float* __restrict__ g3,   const float* __restrict__ be3,
    const float* __restrict__ W4,   const float* __restrict__ b4,
    float* __restrict__ out)
{
  extern __shared__ char smem_raw[];
  __bf16* actBuf   = (__bf16*)smem_raw;                 // 64*1032 bf16 (129KB)
  __bf16* xBuf     = actBuf + ROWS * STRIDE;            // 64*136 bf16 (17KB)
  float*  rowSum   = (float*)(xBuf + ROWS * XS);        // 64
  float*  rowSumSq = rowSum + ROWS;                     // 64
  float*  gbuf     = rowSumSq + ROWS;                   // 1024
  float*  bebuf    = gbuf + HIDDEN;                     // 1024
  float*  red      = bebuf + HIDDEN;                    // 1024

  const int tid = threadIdx.x;
  const int rowBase = blockIdx.x * ROWS;

  // ---- build pair features x = [sat | task | time_emb] as bf16 into xBuf ----
  for (int i = tid; i < ROWS * 128; i += NTHREADS) {
    int m = i >> 7, c = i & 127;
    int r = rowBase + m;
    int b = r >> 13, rem = r & 8191;       // NS*NT = 8192
    int s = rem >> 7, t = rem & 127;       // NT = 128
    float v;
    if (c < 32)       v = sat[(b * 64 + s) * 32 + c];
    else if (c < 64)  v = task[(b * 128 + t) * 32 + (c - 32)];
    else              v = te[time_steps[b] * 64 + (c - 64)];
    xBuf[m * XS + c] = f2bf(v);
  }
  // (mlp_layer_reg's first __syncthreads covers the build)

  mlp_layer_reg(w1s, b1, g1, be1, xBuf,   XS,     actBuf, rowSum, rowSumSq, gbuf, bebuf, 4,  tid);
  mlp_layer_reg(w2s, b2, g2, be2, actBuf, STRIDE, actBuf, rowSum, rowSumSq, gbuf, bebuf, 32, tid);
  mlp_layer_reg(w3s, b3, g3, be3, actBuf, STRIDE, actBuf, rowSum, rowSumSq, gbuf, bebuf, 32, tid);

  // ---- head: 1024 -> 2 (tiny), 8 threads per row, vectorized LDS reads ----
  {
    int row = tid >> 3, part = tid & 7;                 // 8 threads/row, 128 k each
    float a0 = 0.0f, a1 = 0.0f;
    const __bf16* xr = actBuf + row * STRIDE;
    const float2* w4v = (const float2*)W4;              // W4[k][0..1]
    for (int kv = part * 16; kv < part * 16 + 16; ++kv) { // 16 x 8 = 128 k
      v8bf x = *(const v8bf*)(xr + kv * 8);
      for (int e = 0; e < 8; ++e) {
        float xv = bf2f(x[e]);
        float2 w = w4v[kv * 8 + e];
        a0 += xv * w.x;
        a1 += xv * w.y;
      }
    }
    red[(row * 8 + part) * 2 + 0] = a0;
    red[(row * 8 + part) * 2 + 1] = a1;
  }
  __syncthreads();
  if (tid < ROWS) {
    float o0 = b4[0], o1 = b4[1];
    for (int p = 0; p < 8; ++p) {
      o0 += red[(tid * 8 + p) * 2 + 0];
      o1 += red[(tid * 8 + p) * 2 + 1];
    }
    int r = rowBase + tid;
    int b = r >> 13, rem = r & 8191;
    int s = rem >> 7, t = rem & 127;
    bool mk = (cmask[b * 64 + s] != 0) && (tmask[b * 128 + t] != 0);
    out[r]           = (mk ? o0 : -1.0f) * 50.0f;
    out[M_TOTAL + r] = mk ? o1 : -__builtin_inff();
  }
}

extern "C" void kernel_launch(void* const* d_in, const int* in_sizes, int n_in,
                              void* d_out, int out_size, void* d_ws, size_t ws_size,
                              hipStream_t stream) {
  const int*   time_steps = (const int*)d_in[0];
  const float* sat        = (const float*)d_in[1];
  const unsigned char* cm = (const unsigned char*)d_in[2];
  const float* task       = (const float*)d_in[3];
  const unsigned char* tm = (const unsigned char*)d_in[4];
  const float* te         = (const float*)d_in[5];
  const float* W1 = (const float*)d_in[6];
  const float* b1 = (const float*)d_in[7];
  const float* g1 = (const float*)d_in[8];
  const float* be1= (const float*)d_in[9];
  const float* W2 = (const float*)d_in[10];
  const float* b2 = (const float*)d_in[11];
  const float* g2 = (const float*)d_in[12];
  const float* be2= (const float*)d_in[13];
  const float* W3 = (const float*)d_in[14];
  const float* b3 = (const float*)d_in[15];
  const float* g3 = (const float*)d_in[16];
  const float* be3= (const float*)d_in[17];
  const float* W4 = (const float*)d_in[18];
  const float* b4 = (const float*)d_in[19];

  __bf16* w1s = (__bf16*)d_ws;                // 128*1024  bf16 = 256KB
  __bf16* w2s = w1s + 128 * 1024;             // 1024*1024 bf16 = 2MB
  __bf16* w3s = w2s + 1024 * 1024;            // 1024*1024 bf16 = 2MB

  swizzle_w<<<(128 * 1024 + 255) / 256, 256, 0, stream>>>(W1, w1s, 128, 1024);
  swizzle_w<<<(1024 * 1024 + 255) / 256, 256, 0, stream>>>(W2, w2s, 1024, 1024);
  swizzle_w<<<(1024 * 1024 + 255) / 256, 256, 0, stream>>>(W3, w3s, 1024, 1024);

  size_t shmem = (size_t)(ROWS * STRIDE * 2)                      // act buffer
               + (size_t)(ROWS * XS * 2)                          // feature buffer
               + (size_t)(2 * ROWS + 2 * HIDDEN + 1024) * 4;      // stats/g/be/red
  fused_mlp<<<M_TOTAL / ROWS, NTHREADS, shmem, stream>>>(
      time_steps, sat, cm, task, tm, te,
      w1s, b1, g1, be1, w2s, b2, g2, be2, w3s, b3, g3, be3,
      W4, b4, (float*)d_out);
}
